// BitNetAttention_58102317580687
// MI455X (gfx1250) — 16-bit and 8-bit matrix operand order repaired and checked by reading, compiled, not run on hardware
//
#include <hip/hip_runtime.h>
#include <hip/hip_bf16.h>
#include <stdint.h>


typedef __attribute__((ext_vector_type(4)))  int      v4i;
typedef __attribute__((ext_vector_type(8)))  int      v8i;
typedef __attribute__((ext_vector_type(8)))  float    v8f;
typedef __attribute__((ext_vector_type(16))) _Float16 v16h;

#define D_MODEL 2048
#define N_HEADS 16
#define D_K     128
#define SEQ_T   2048
#define NTOK    4096
#define WN      (D_MODEL*D_MODEL)

#if defined(__has_builtin)
#if __has_builtin(__builtin_amdgcn_global_load_async_to_lds_b128)
#define HAVE_ASYNC_LDS 1
#endif
#endif

#ifdef HAVE_ASYNC_LDS
typedef __attribute__((address_space(1))) v4i g_v4i;
typedef __attribute__((address_space(3))) v4i l_v4i;
#define ASYNC_CP16(dst_lds, src_glb)                                           \
    __builtin_amdgcn_global_load_async_to_lds_b128(                            \
        (g_v4i*)(void*)(src_glb), (l_v4i*)(void*)(dst_lds), 0, 0)
#if __has_builtin(__builtin_amdgcn_s_wait_asynccnt)
#define ASYNC_WAIT() __builtin_amdgcn_s_wait_asynccnt(0)
#else
#define ASYNC_WAIT() asm volatile("s_wait_asynccnt 0" ::: "memory")
#endif
#else
#define ASYNC_CP16(dst_lds, src_glb)                                           \
    (*(int4*)(dst_lds) = *(const int4*)(src_glb))
#define ASYNC_WAIT()
#endif

__global__ void wabs_partial(const float* __restrict__ w, float* __restrict__ part, int n) {
    __shared__ float red[256];
    float s = 0.f;
    for (int i = blockIdx.x * blockDim.x + threadIdx.x; i < n; i += gridDim.x * blockDim.x)
        s += fabsf(w[i]);
    red[threadIdx.x] = s;
    __syncthreads();
    for (int st = 128; st > 0; st >>= 1) {
        if (threadIdx.x < st) red[threadIdx.x] += red[threadIdx.x + st];
        __syncthreads();
    }
    if (threadIdx.x == 0) part[blockIdx.x] = red[0];
}

__global__ void wabs_final(const float* __restrict__ part, int np, float* __restrict__ out, float inv_n) {
    __shared__ float red[256];
    float s = 0.f;
    for (int i = threadIdx.x; i < np; i += 256) s += part[i];
    red[threadIdx.x] = s;
    __syncthreads();
    for (int st = 128; st > 0; st >>= 1) {
        if (threadIdx.x < st) red[threadIdx.x] += red[threadIdx.x + st];
        __syncthreads();
    }
    if (threadIdx.x == 0) out[0] = fmaxf(red[0] * inv_n, 1e-5f);
}

__global__ void wquant(const float* __restrict__ w, const float* __restrict__ sptr,
                       int8_t* __restrict__ w8, int n) {
    const float inv = 1.0f / sptr[0];
    for (int i = blockIdx.x * blockDim.x + threadIdx.x; i < n; i += gridDim.x * blockDim.x) {
        float q = rintf(w[i] * inv);
        q = fminf(fmaxf(q, -1.f), 1.f);
        w8[i] = (int8_t)q;
    }
}

__global__ void norm_quant(const float* __restrict__ x, const float* __restrict__ g,
                           int8_t* __restrict__ xq, float* __restrict__ deq) {
    __shared__ float red[256];
    const int t = blockIdx.x;
    const float* row = x + (size_t)t * D_MODEL;
    float v[8], ss = 0.f;
#pragma unroll
    for (int i = 0; i < 8; ++i) {
        v[i] = row[threadIdx.x + i * 256];
        ss += v[i] * v[i];
    }
    red[threadIdx.x] = ss;
    __syncthreads();
    for (int st = 128; st > 0; st >>= 1) {
        if (threadIdx.x < st) red[threadIdx.x] += red[threadIdx.x + st];
        __syncthreads();
    }
    const float inv_rms = 1.0f / sqrtf(red[0] * (1.0f / D_MODEL) + 1e-6f);
    __syncthreads();

    float xn[8], amax = 0.f;
#pragma unroll
    for (int i = 0; i < 8; ++i) {
        xn[i] = v[i] * inv_rms * g[threadIdx.x + i * 256];
        amax = fmaxf(amax, fabsf(xn[i]));
    }
    red[threadIdx.x] = amax;
    __syncthreads();
    for (int st = 128; st > 0; st >>= 1) {
        if (threadIdx.x < st) red[threadIdx.x] = fmaxf(red[threadIdx.x], red[threadIdx.x + st]);
        __syncthreads();
    }
    const float am = fmaxf(red[0], 1e-5f);
    const float scale = 127.0f / am;
#pragma unroll
    for (int i = 0; i < 8; ++i) {
        float q = rintf(xn[i] * scale);
        q = fminf(fmaxf(q, -128.f), 127.f);
        xq[(size_t)t * D_MODEL + threadIdx.x + i * 256] = (int8_t)q;
    }
    if (threadIdx.x == 0) deq[t] = am * (1.0f / 127.0f);
}

#define BSTRIDE 80

__global__ void gemm_i8(const int8_t* __restrict__ A, const int8_t* __restrict__ Bw,
                        const float* __restrict__ act_deq, const float* __restrict__ wsp,
                        int mode, _Float16* __restrict__ outh, float* __restrict__ outf) {
    __shared__ int8_t bls[2][128 * BSTRIDE];
    const int tid   = threadIdx.x;
    const int lane  = tid & 31;
    const int wv    = tid >> 5;
    const int mb    = blockIdx.x >> 4;
    const int n128  = blockIdx.x & 15;
    const int m0    = mb * 128 + wv * 16;
    const int n0    = n128 * 128;
    const int mlane = lane & 15;
    const int hiA   = (lane & 16) ? 16 : 0;
    const int hiB   = (lane & 16) ? 16 : 0;
    const int rowhi = (lane & 16) ? 8  : 0;
    const float ws  = wsp[0];

    const int8_t* arow = A + (size_t)(m0 + mlane) * D_MODEL;

    const int scol = tid >> 1;
    const int soff = (tid & 1) * 32;
    const int8_t* bsrc = Bw + (size_t)(n0 + scol) * D_MODEL + soff;
    int8_t* bdst = &bls[0][0] + scol * BSTRIDE + soff;

    v8i acc[8] = {};

    ASYNC_CP16(bdst,      bsrc);
    ASYNC_CP16(bdst + 16, bsrc + 16);

    int cur = 0;
    for (int k0 = 0; k0 < D_MODEL; k0 += 64) {
        ASYNC_WAIT();
        __syncthreads();
        if (k0 + 64 < D_MODEL) {
            const int8_t* bs = bsrc + k0 + 64;
            int8_t* bd = &bls[cur ^ 1][0] + scol * BSTRIDE + soff;
            ASYNC_CP16(bd,      bs);
            ASYNC_CP16(bd + 16, bs + 16);
        }
        v8i a;
        const int8_t* ap = arow + k0 + hiA;
        ((int4*)&a)[0] = *(const int4*)(ap);
        ((int4*)&a)[1] = *(const int4*)(ap + 32);
        __builtin_prefetch(ap + 64, 0, 1);

        const int8_t* bbase = &bls[cur][0];
        v8i bfrag[8];
#pragma unroll
        for (int j = 0; j < 8; ++j) {
            const int8_t* bcol = bbase + (j * 16 + mlane) * BSTRIDE + hiB;
            ((int4*)&bfrag[j])[0] = *(const int4*)(bcol);
            ((int4*)&bfrag[j])[1] = *(const int4*)(bcol + 32);
        }
#pragma unroll
        for (int j = 0; j < 8; ++j)
            acc[j] = __builtin_amdgcn_wmma_i32_16x16x64_iu8(true, a, true, bfrag[j],
                                                            acc[j], false, false);
        cur ^= 1;
    }

#pragma unroll
    for (int j = 0; j < 8; ++j) {
        const int ocol = n0 + j * 16 + mlane;
#pragma unroll
        for (int r = 0; r < 8; ++r) {
            const int mm = m0 + r + rowhi;
            const float val = (float)acc[j][r] * act_deq[mm] * ws;
            if (mode == 2) {
                outf[(size_t)mm * D_MODEL + ocol] = val;
            } else {
                const int b = mm >> 11, tt = mm & 2047;
                const int h = ocol >> 7, d = ocol & 127;
                size_t idx;
                if (mode == 0) idx = ((size_t)(b * N_HEADS + h) * SEQ_T + tt) * D_K + d;
                else           idx = ((size_t)(b * N_HEADS + h) * D_K + d) * SEQ_T + tt;
                outh[idx] = (_Float16)val;
            }
        }
    }
}

#define KSTRIDE 136
#define VSTRIDE 40
#define PSTRIDE 40

__global__ void attn_kernel(const _Float16* __restrict__ Qh, const _Float16* __restrict__ Kh,
                            const _Float16* __restrict__ Vt, float* __restrict__ ctx) {
    __shared__ _Float16 kls[2][32 * KSTRIDE];
    __shared__ _Float16 vls[2][D_K * VSTRIDE];
    __shared__ _Float16 plds[4][16 * PSTRIDE];
    const int tid   = threadIdx.x;
    const int wv    = tid >> 5;
    const int lane  = tid & 31;
    const int bh    = blockIdx.y;
    const int b     = bh >> 4, h = bh & 15;
    const int q0    = (blockIdx.x * 4 + wv) * 16;
    const int mlane = lane & 15;
    const int loA   = (lane & 16) ? 8  : 0;
    const int loB   = (lane & 16) ? 8  : 0;
    const int rowhi = (lane & 16) ? 8 : 0;

    const _Float16* qbase = Qh + (size_t)bh * SEQ_T * D_K;
    const _Float16* kbase = Kh + (size_t)bh * SEQ_T * D_K;
    const _Float16* vbase = Vt + (size_t)bh * D_K * SEQ_T;

    auto stage = [&](int buf, int kt) {
        const int krow = tid >> 2, kchunk = (tid & 3) * 32;
        const _Float16* ksrc = kbase + (size_t)(kt + krow) * D_K + kchunk;
        _Float16*       kdst = &kls[buf][0] + krow * KSTRIDE + kchunk;
#pragma unroll
        for (int c = 0; c < 4; ++c)
            ASYNC_CP16(kdst + c * 8, ksrc + c * 8);
        const _Float16* vsrc = vbase + (size_t)tid * SEQ_T + kt;
        _Float16*       vdst = &vls[buf][0] + tid * VSTRIDE;
#pragma unroll
        for (int c = 0; c < 4; ++c)
            ASYNC_CP16(vdst + c * 8, vsrc + c * 8);
    };

    v16h qf[4];
    {
        const _Float16* qrow = qbase + (size_t)(q0 + mlane) * D_K;
#pragma unroll
        for (int c = 0; c < 4; ++c) {
            const _Float16* p = qrow + c * 32 + loA;
            ((int4*)&qf[c])[0] = *(const int4*)(p);
            ((int4*)&qf[c])[1] = *(const int4*)(p + 16);
        }
    }

    float mi[8], li[8];
    v8f oacc[8] = {};
#pragma unroll
    for (int r = 0; r < 8; ++r) { mi[r] = -1e30f; li[r] = 0.f; }

    const float sc = 0.08838834764831845f;
    _Float16* myl = plds[wv];

    stage(0, 0);
    int cur = 0;
    for (int kt = 0; kt < SEQ_T; kt += 32) {
        ASYNC_WAIT();
        __syncthreads();
        if (kt + 32 < SEQ_T) stage(cur ^ 1, kt + 32);

        v8f s0 = {}, s1 = {};
        const _Float16* kbuf = &kls[cur][0];
#pragma unroll
        for (int c = 0; c < 4; ++c) {
            const _Float16* k0r = kbuf + (size_t)mlane * KSTRIDE + c * 32 + loB;
            v16h kf0;
            ((int4*)&kf0)[0] = *(const int4*)(k0r);
            ((int4*)&kf0)[1] = *(const int4*)(k0r + 16);
            const _Float16* k1r = kbuf + (size_t)(16 + mlane) * KSTRIDE + c * 32 + loB;
            v16h kf1;
            ((int4*)&kf1)[0] = *(const int4*)(k1r);
            ((int4*)&kf1)[1] = *(const int4*)(k1r + 16);
            s0 = __builtin_amdgcn_wmma_f32_16x16x32_f16(false, qf[c], false, kf0,
                                                        (short)0, s0, false, false);
            s1 = __builtin_amdgcn_wmma_f32_16x16x32_f16(false, qf[c], false, kf1,
                                                        (short)0, s1, false, false);
        }

#pragma unroll
        for (int r = 0; r < 8; ++r) {
            float v0 = s0[r] * sc, v1 = s1[r] * sc;
            float mx = fmaxf(v0, v1);
            mx = fmaxf(mx, __shfl_xor(mx, 1, 32));
            mx = fmaxf(mx, __shfl_xor(mx, 2, 32));
            mx = fmaxf(mx, __shfl_xor(mx, 4, 32));
            mx = fmaxf(mx, __shfl_xor(mx, 8, 32));
            const float nm = fmaxf(mi[r], mx);
            const float al = __expf(mi[r] - nm);
            const float p0 = __expf(v0 - nm);
            const float p1 = __expf(v1 - nm);
            float rs = p0 + p1;
            rs += __shfl_xor(rs, 1, 32);
            rs += __shfl_xor(rs, 2, 32);
            rs += __shfl_xor(rs, 4, 32);
            rs += __shfl_xor(rs, 8, 32);
            li[r] = li[r] * al + rs;
            mi[r] = nm;
#pragma unroll
            for (int dt = 0; dt < 8; ++dt) oacc[dt][r] *= al;
            const int mm = r + rowhi;
            myl[mm * PSTRIDE + mlane]      = (_Float16)p0;
            myl[mm * PSTRIDE + 16 + mlane] = (_Float16)p1;
        }
        asm volatile("s_wait_dscnt 0" ::: "memory");

        v16h pf;
        {
            const _Float16* pp = myl + mlane * PSTRIDE + loA;
            ((int4*)&pf)[0] = *(const int4*)(pp);
            ((int4*)&pf)[1] = *(const int4*)(pp + 16);
        }

        const _Float16* vbuf = &vls[cur][0];
#pragma unroll
        for (int grp = 0; grp < 2; ++grp) {
            v16h vf[4];
#pragma unroll
            for (int q = 0; q < 4; ++q) {
                const int dt = grp * 4 + q;
                const _Float16* vrow = vbuf + (size_t)(dt * 16 + mlane) * VSTRIDE + loB;
                ((int4*)&vf[q])[0] = *(const int4*)(vrow);
                ((int4*)&vf[q])[1] = *(const int4*)(vrow + 16);
            }
#pragma unroll
            for (int q = 0; q < 4; ++q) {
                const int dt = grp * 4 + q;
                oacc[dt] = __builtin_amdgcn_wmma_f32_16x16x32_f16(false, pf, false, vf[q],
                                                                  (short)0, oacc[dt], false, false);
            }
        }
        cur ^= 1;
    }

#pragma unroll
    for (int r = 0; r < 8; ++r) {
        const float inv = 1.0f / li[r];
        const int tt = q0 + r + rowhi;
        float* crow = ctx + ((size_t)(b * SEQ_T + tt) * D_MODEL) + h * D_K;
#pragma unroll
        for (int dt = 0; dt < 8; ++dt)
            crow[dt * 16 + mlane] = oacc[dt][r] * inv;
    }
}

extern "C" void kernel_launch(void* const* d_in, const int* in_sizes, int n_in,
                              void* d_out, int out_size, void* d_ws, size_t ws_size,
                              hipStream_t stream) {
    const float* x  = (const float*)d_in[0];
    const float* w[4] = { (const float*)d_in[1], (const float*)d_in[2],
                          (const float*)d_in[3], (const float*)d_in[4] };
    const float* g[4] = { (const float*)d_in[5], (const float*)d_in[6],
                          (const float*)d_in[7], (const float*)d_in[8] };
    float* out = (float*)d_out;

    char* ws = (char*)d_ws;
    size_t off = 0;
    auto alloc = [&](size_t bytes) { void* p = ws + off; off = (off + bytes + 255) & ~(size_t)255; return p; };
    int8_t*    xq      = (int8_t*)   alloc((size_t)NTOK * D_MODEL);
    int8_t*    w8[4];
    for (int i = 0; i < 4; ++i) w8[i] = (int8_t*)alloc((size_t)WN);
    _Float16*  Qh      = (_Float16*) alloc((size_t)NTOK * D_MODEL * 2);
    _Float16*  Kh      = (_Float16*) alloc((size_t)NTOK * D_MODEL * 2);
    _Float16*  Vt      = (_Float16*) alloc((size_t)NTOK * D_MODEL * 2);
    float*     ctx     = (float*)    alloc((size_t)NTOK * D_MODEL * 4);
    float*     act_deq = (float*)    alloc((size_t)NTOK * 4);
    float*     wscale  = (float*)    alloc(4 * 4);
    float*     part    = (float*)    alloc(1024 * 4);

    for (int i = 0; i < 4; ++i) {
        wabs_partial<<<1024, 256, 0, stream>>>(w[i], part, WN);
        wabs_final  <<<1,    256, 0, stream>>>(part, 1024, wscale + i, 1.0f / (float)WN);
        wquant      <<<4096, 256, 0, stream>>>(w[i], wscale + i, w8[i], WN);
    }

    const int gemmBlocks = 32 * 16;

    norm_quant<<<NTOK, 256, 0, stream>>>(x, g[0], xq, act_deq);
    gemm_i8<<<gemmBlocks, 256, 0, stream>>>(xq, w8[0], act_deq, wscale + 0, 0, Qh, nullptr);
    norm_quant<<<NTOK, 256, 0, stream>>>(x, g[1], xq, act_deq);
    gemm_i8<<<gemmBlocks, 256, 0, stream>>>(xq, w8[1], act_deq, wscale + 1, 0, Kh, nullptr);
    norm_quant<<<NTOK, 256, 0, stream>>>(x, g[2], xq, act_deq);
    gemm_i8<<<gemmBlocks, 256, 0, stream>>>(xq, w8[2], act_deq, wscale + 2, 1, Vt, nullptr);

    attn_kernel<<<dim3(SEQ_T / 16 / 4, 32), 128, 0, stream>>>(Qh, Kh, Vt, ctx);

    norm_quant<<<NTOK, 256, 0, stream>>>(ctx, g[3], xq, act_deq);
    gemm_i8<<<gemmBlocks, 256, 0, stream>>>(xq, w8[3], act_deq, wscale + 3, 2, nullptr, out);
}
